// CustomBiLSTM_55267639165093
// MI455X (gfx1250) — matrix operand order repaired and checked by reading, not compiled, not run on hardware
//
#include <hip/hip_runtime.h>

#define T_STEPS 512
#define BATCH   64
#define HDIM    256
#define GDIM    1024
#define WPITCH  144   // bf16 elems per LDS weight row (128 cols + 16 pad, keeps 32B align, skews banks)
#define GPITCH  132   // f32 elems per LDS gate row (128 + 4 pad)
#define NWG_DIR 8

typedef __bf16 v16bf __attribute__((ext_vector_type(16)));
typedef float  v8f   __attribute__((ext_vector_type(8)));

struct BFrag { uint4 lo, hi; };

__device__ __forceinline__ unsigned short f2bf(float x) {
  unsigned u = __builtin_bit_cast(unsigned, x);
  unsigned r = u + 0x7FFFu + ((u >> 16) & 1u);  // round-to-nearest-even
  return (unsigned short)(r >> 16);
}

__device__ __forceinline__ float sigm(float x) {
  return 1.0f / (1.0f + __expf(-x));
}
__device__ __forceinline__ float tanh_fast(float x) {
  float e = __expf(-2.0f * fabsf(x));
  float t = (1.0f - e) / (1.0f + e);
  return x >= 0.0f ? t : -t;
}

// Monotonic-phase global barrier across the 8 workgroups of one direction.
__device__ __forceinline__ void gbar(unsigned* cnt, unsigned target) {
  __threadfence();
  __syncthreads();
  if (threadIdx.x == 0) {
    __hip_atomic_fetch_add(cnt, 1u, __ATOMIC_RELEASE, __HIP_MEMORY_SCOPE_AGENT);
    while (__hip_atomic_load(cnt, __ATOMIC_RELAXED, __HIP_MEMORY_SCOPE_AGENT) < target) {
      __builtin_amdgcn_s_sleep(2);
    }
  }
  __threadfence();
  __syncthreads();
}

__global__ void init_cnt_kernel(unsigned* c) {
  if (threadIdx.x < 8) c[threadIdx.x] = 0u;
}

__global__ void f32_to_bf16_kernel(const float* __restrict__ src,
                                   unsigned short* __restrict__ dst, int n) {
  int i = blockIdx.x * 256 + threadIdx.x;
  if (i < n) dst[i] = f2bf(src[i]);
}

// Build combined bf16 weight matrix Wg[k][col] (k<256 -> Whh rows, else Wih rows).
__global__ void prep_w_kernel(const float* __restrict__ Wih, const float* __restrict__ Whh,
                              int K, unsigned short* __restrict__ dst) {
  int i = blockIdx.x * 256 + threadIdx.x;
  if (i < K * GDIM) {
    int k = i >> 10, col = i & 1023;
    float v = (k < HDIM) ? Whh[(size_t)k * GDIM + col] : Wih[(size_t)(k - HDIM) * GDIM + col];
    dst[i] = f2bf(v);
  }
}

// One bidirectional LSTM layer, fused projection: gates = [h | x_t] @ [Whh;Wih] + b
// grid = 16 blocks (blockIdx&1 = direction, blockIdx>>1 = H-slice of 32), 256 threads (8 waves).
__global__ void __launch_bounds__(256, 1)
lstm_layer_kernel(const unsigned short* __restrict__ xbf, int Ix, int K,
                  const unsigned short* __restrict__ Wg,      // [2][K][1024] bf16
                  const float* __restrict__ bias_f, const float* __restrict__ bias_b,
                  const long long* __restrict__ lengths,
                  unsigned short* __restrict__ hbuf,          // [2 dir][2 phase][64][256] bf16
                  unsigned short* __restrict__ out_bf,        // layer0: (T,B,512) bf16, else null
                  float* __restrict__ out_f32,                // layer1: (T,B,512) f32, else null
                  float* __restrict__ hn, float* __restrict__ cn,
                  unsigned* __restrict__ counters) {
  extern __shared__ __attribute__((aligned(32))) char smem[];
  unsigned short* Wlds = (unsigned short*)smem;                    // K * WPITCH bf16
  float* gates = (float*)(smem + (size_t)K * WPITCH * 2);          // 64 * GPITCH f32
  float* biasL = gates + BATCH * GPITCH;                           // 128 f32

  const int tid  = threadIdx.x;
  const int lane = tid & 31;
  const int wave = tid >> 5;
  const int dir  = blockIdx.x & 1;
  const int wg   = blockIdx.x >> 1;      // 0..7
  const int jc   = wg * 32;              // owned H columns [jc, jc+32)
  const float* bias = dir ? bias_b : bias_f;
  const unsigned short* Wsrc = Wg + (size_t)dir * K * GDIM;
  unsigned short* hb = hbuf + (size_t)dir * 2 * BATCH * HDIM;
  unsigned* cnt = counters + dir;

  // Weight slice -> LDS via CDNA5 async memory->LDS DMA (no VGPR staging, ASYNCcnt-tracked).
  // Per k-row, the 4 gate chunks of 32 cols; K*16 is a multiple of 256 so EXEC is full.
  for (int idx = tid; idx < K * 16; idx += 256) {
    int k = idx >> 4, part = idx & 15;
    int chunk = part >> 2, sub = part & 3;
    const unsigned short* gsrc = Wsrc + (size_t)k * GDIM + chunk * 256 + jc + sub * 8;
    unsigned ldsoff = (unsigned)(uintptr_t)(Wlds + (size_t)k * WPITCH + chunk * 32 + sub * 8);
    asm volatile("global_load_async_to_lds_b128 %0, %1, off"
                 :: "v"(ldsoff), "v"(gsrc)
                 : "memory");
  }
  asm volatile("s_wait_asynccnt 0x0" ::: "memory");

  if (tid < 128) {
    biasL[tid] = bias[(tid >> 5) * 256 + jc + (tid & 31)];
  }
  {
    // zero both phases of our h slice
    int b = tid & 63, j = (tid >> 6) * 8;
    uint4 z = make_uint4(0u, 0u, 0u, 0u);
    *(uint4*)(hb + (size_t)b * HDIM + jc + j) = z;
    *(uint4*)(hb + (size_t)BATCH * HDIM + (size_t)b * HDIM + jc + j) = z;
  }

  const int b_e = tid & 63;           // batch row owned for elementwise
  const int jb  = (tid >> 6) * 8;     // 8 H-cols owned within slice
  float hreg[8], creg[8];
#pragma unroll
  for (int q = 0; q < 8; ++q) { hreg[q] = 0.0f; creg[q] = 0.0f; }
  const int mylen = (int)lengths[b_e];

  gbar(cnt, NWG_DIR * 1u);

  // WMMA tile assignment: wave -> m-tile (batch rows) + 4 n-tiles (gate cols)
  const int m    = wave >> 1;
  const int nh   = (wave & 1) * 4;
  const int arow = 16 * m + (lane & 15);   // A-fragment row (batch)
  const int asub = (lane >> 4) * 8;        // per-lane K sub-chunk offset (0 or 8)
  const int NK   = K >> 5;

  for (int s = 0; s < T_STEPS; ++s) {
    const int t = dir ? (T_STEPS - 1 - s) : s;
    const unsigned short* hrow = hb + (size_t)(s & 1) * BATCH * HDIM + (size_t)arow * HDIM;
    const unsigned short* xrow = xbf + (size_t)t * BATCH * Ix + (size_t)arow * Ix;

    if (s + 1 < T_STEPS) {
      int tn = dir ? (T_STEPS - 2 - s) : (s + 1);
      __builtin_prefetch(xbf + (size_t)tn * BATCH * Ix + (size_t)arow * Ix, 0, 1);
    }

    v8f acc0 = {}, acc1 = {}, acc2 = {}, acc3 = {};
    for (int kc = 0; kc < NK; ++kc) {
      const int k0 = kc * 32 + asub;
      // A = [h | x_t]; chunks never straddle the h/x boundary (multiple of 32)
      const unsigned short* src = (k0 < HDIM) ? (hrow + k0) : (xrow + (k0 - HDIM));
      BFrag af;
      af.lo = *(const uint4*)(src);        // K = k0 .. k0+7
      af.hi = *(const uint4*)(src + 16);   // K = k0+16 .. k0+23
      v16bf a = __builtin_bit_cast(v16bf, af);
      const unsigned short* wrow = Wlds + (size_t)(kc * 32 + asub) * WPITCH + nh * 16 + (lane & 15);
      v16bf b0, b1, b2, b3;
#pragma unroll
      for (int e = 0; e < 16; ++e) {
        const unsigned short* p = wrow + (size_t)((e < 8) ? e : (e + 8)) * WPITCH;
        b0[e] = __builtin_bit_cast(__bf16, p[0]);
        b1[e] = __builtin_bit_cast(__bf16, p[16]);
        b2[e] = __builtin_bit_cast(__bf16, p[32]);
        b3[e] = __builtin_bit_cast(__bf16, p[48]);
      }
      acc0 = __builtin_amdgcn_wmma_f32_16x16x32_bf16(false, a, false, b0, (short)0, acc0, false, false);
      acc1 = __builtin_amdgcn_wmma_f32_16x16x32_bf16(false, a, false, b1, (short)0, acc1, false, false);
      acc2 = __builtin_amdgcn_wmma_f32_16x16x32_bf16(false, a, false, b2, (short)0, acc2, false, false);
      acc3 = __builtin_amdgcn_wmma_f32_16x16x32_bf16(false, a, false, b3, (short)0, acc3, false, false);
    }

    // Stage gate tiles (+bias) into LDS for the cross-wave elementwise phase.
    {
      const int colb = lane & 15;
      const int rowb = 16 * m + asub;   // D layout: M = (lane>>4)*8 + r
#pragma unroll
      for (int j = 0; j < 4; ++j) {
        v8f accv = (j == 0) ? acc0 : (j == 1) ? acc1 : (j == 2) ? acc2 : acc3;
        int col = (nh + j) * 16 + colb;
        float bv = biasL[col];
#pragma unroll
        for (int r = 0; r < 8; ++r)
          gates[(rowb + r) * GPITCH + col] = accv[r] + bv;
      }
    }
    __syncthreads();

    // Elementwise LSTM cell for (b_e, jc+jb .. jc+jb+7)
    float outv[8];
    const bool mk = (t < mylen);
#pragma unroll
    for (int q = 0; q < 8; ++q) {
      int j = jb + q;
      float gi = gates[b_e * GPITCH + j];
      float gf = gates[b_e * GPITCH + 32 + j];
      float go = gates[b_e * GPITCH + 64 + j];
      float gg = gates[b_e * GPITCH + 96 + j];
      float c1 = sigm(gf + 1.0f) * creg[q] + sigm(gi) * tanh_fast(gg);
      float h1 = sigm(go) * tanh_fast(c1);
      hreg[q] = mk ? h1 : hreg[q];
      creg[q] = mk ? c1 : creg[q];
      outv[q] = mk ? h1 : 0.0f;
    }
    // Publish next-phase h (bf16, double-buffered so other WGs can still read phase s)
    {
      unsigned short* hw = hb + (size_t)((s + 1) & 1) * BATCH * HDIM + (size_t)b_e * HDIM + jc + jb;
      uint4 hp;
      hp.x = (unsigned)f2bf(hreg[0]) | ((unsigned)f2bf(hreg[1]) << 16);
      hp.y = (unsigned)f2bf(hreg[2]) | ((unsigned)f2bf(hreg[3]) << 16);
      hp.z = (unsigned)f2bf(hreg[4]) | ((unsigned)f2bf(hreg[5]) << 16);
      hp.w = (unsigned)f2bf(hreg[6]) | ((unsigned)f2bf(hreg[7]) << 16);
      *(uint4*)hw = hp;
    }
    // Sequence output written at scan index s (reproduces backward scan-order stacking)
    {
      const size_t obase = (size_t)s * BATCH * 512 + (size_t)b_e * 512 + (size_t)dir * 256 + jc + jb;
      if (out_f32) {
        *(float4*)(out_f32 + obase)     = make_float4(outv[0], outv[1], outv[2], outv[3]);
        *(float4*)(out_f32 + obase + 4) = make_float4(outv[4], outv[5], outv[6], outv[7]);
      } else {
        uint4 op;
        op.x = (unsigned)f2bf(outv[0]) | ((unsigned)f2bf(outv[1]) << 16);
        op.y = (unsigned)f2bf(outv[2]) | ((unsigned)f2bf(outv[3]) << 16);
        op.z = (unsigned)f2bf(outv[4]) | ((unsigned)f2bf(outv[5]) << 16);
        op.w = (unsigned)f2bf(outv[6]) | ((unsigned)f2bf(outv[7]) << 16);
        *(uint4*)(out_bf + obase) = op;
      }
    }
    gbar(cnt, NWG_DIR * (unsigned)(s + 2));
  }

  // Final carries hN / cN
  {
    float* hdst = hn + (size_t)dir * BATCH * HDIM + (size_t)b_e * HDIM + jc + jb;
    float* cdst = cn + (size_t)dir * BATCH * HDIM + (size_t)b_e * HDIM + jc + jb;
#pragma unroll
    for (int q = 0; q < 8; ++q) { hdst[q] = hreg[q]; cdst[q] = creg[q]; }
  }
}

extern "C" void kernel_launch(void* const* d_in, const int* in_sizes, int n_in,
                              void* d_out, int out_size, void* d_ws, size_t ws_size,
                              hipStream_t stream) {
  (void)in_sizes; (void)n_in; (void)out_size; (void)ws_size;
  const float* x_in        = (const float*)d_in[0];
  const long long* lengths = (const long long*)d_in[1];
  const float* Wih_f0 = (const float*)d_in[2];
  const float* Whh_f0 = (const float*)d_in[3];
  const float* b_f0   = (const float*)d_in[4];
  const float* Wih_b0 = (const float*)d_in[5];
  const float* Whh_b0 = (const float*)d_in[6];
  const float* b_b0   = (const float*)d_in[7];
  const float* Wih_f1 = (const float*)d_in[8];
  const float* Whh_f1 = (const float*)d_in[9];
  const float* b_f1   = (const float*)d_in[10];
  const float* Wih_b1 = (const float*)d_in[11];
  const float* Whh_b1 = (const float*)d_in[12];
  const float* b_b1   = (const float*)d_in[13];
  float* out = (float*)d_out;

  char* ws = (char*)d_ws;
  unsigned*       cnt  = (unsigned*)ws;                                        // 8 u32 (layer0: 0..1, layer1: 2..3)
  unsigned short* xbf0 = (unsigned short*)(ws + 256);                          // (T,B,256) bf16
  unsigned short* xbf1 = (unsigned short*)(ws + 256 + 16777216);               // (T,B,512) bf16
  unsigned short* w0   = (unsigned short*)(ws + 256 + 16777216 + 33554432);    // [2][512][1024] bf16
  unsigned short* w1   = (unsigned short*)((char*)w0 + (size_t)2 * 512 * 1024 * 2);  // [2][768][1024]
  unsigned short* hbuf = (unsigned short*)((char*)w1 + (size_t)2 * 768 * 1024 * 2);  // [2][2][64][256]

  init_cnt_kernel<<<1, 32, 0, stream>>>(cnt);
  {
    int n = T_STEPS * BATCH * 256;
    f32_to_bf16_kernel<<<(n + 255) / 256, 256, 0, stream>>>(x_in, xbf0, n);
  }
  {
    int n0 = 512 * GDIM, n1 = 768 * GDIM;
    prep_w_kernel<<<(n0 + 255) / 256, 256, 0, stream>>>(Wih_f0, Whh_f0, 512, w0);
    prep_w_kernel<<<(n0 + 255) / 256, 256, 0, stream>>>(Wih_b0, Whh_b0, 512, w0 + (size_t)512 * GDIM);
    prep_w_kernel<<<(n1 + 255) / 256, 256, 0, stream>>>(Wih_f1, Whh_f1, 768, w1);
    prep_w_kernel<<<(n1 + 255) / 256, 256, 0, stream>>>(Wih_b1, Whh_b1, 768, w1 + (size_t)768 * GDIM);
  }

  (void)hipFuncSetAttribute(reinterpret_cast<const void*>(&lstm_layer_kernel),
                            hipFuncAttributeMaxDynamicSharedMemorySize, 262144);

  const size_t OUT_SZ = (size_t)T_STEPS * BATCH * 512;
  float* hn_base = out + OUT_SZ;
  float* cn_base = out + OUT_SZ + (size_t)4 * BATCH * HDIM;

  const size_t smem0 = (size_t)512 * WPITCH * 2 + (size_t)BATCH * GPITCH * 4 + 512;  // 181,760 B
  const size_t smem1 = (size_t)768 * WPITCH * 2 + (size_t)BATCH * GPITCH * 4 + 512;  // 255,488 B

  lstm_layer_kernel<<<16, 256, smem0, stream>>>(
      xbf0, 256, 512, w0, b_f0, b_b0, lengths, hbuf,
      xbf1, nullptr, hn_base, cn_base, cnt);

  lstm_layer_kernel<<<16, 256, smem1, stream>>>(
      xbf1, 512, 768, w1, b_f1, b_b1, lengths, hbuf,
      nullptr, out, hn_base + (size_t)2 * BATCH * HDIM, cn_base + (size_t)2 * BATCH * HDIM, cnt + 2);
}
